// TreeGRUConv_11304353923841
// MI455X (gfx1250) — compile-verified, matrix operand order checked by reading, not run on hardware
//
#include <hip/hip_runtime.h>

typedef __attribute__((ext_vector_type(16))) __bf16 v16bf;
typedef __attribute__((ext_vector_type(8)))  float  v8f;

#define FDIM 128
#define HDIM 256
#define THDIM 768        // 3*H
#define FPAD (FDIM + 8)  // padded LDS row (bf16 elems) for x tile
#define HPAD (HDIM + 8)  // padded LDS row (bf16 elems) for h tile
#define MROWS 32         // rows per block (2 x 16 register tiles in M)

// ---------------------------------------------------------------------------
// Zero-fill (graph-capture safe)
// ---------------------------------------------------------------------------
__global__ void zero_f32(float4* __restrict__ p, long n4) {
    long i = (long)blockIdx.x * blockDim.x + threadIdx.x;
    if (i < n4) p[i] = make_float4(0.f, 0.f, 0.f, 0.f);
}

// ---------------------------------------------------------------------------
// One-shot f32 -> bf16 weight conversion into workspace
// ---------------------------------------------------------------------------
__global__ void cvt_weights(const float* __restrict__ wih, const float* __restrict__ whh,
                            __bf16* __restrict__ bih, __bf16* __restrict__ bhh) {
    int i = blockIdx.x * blockDim.x + threadIdx.x;
    if (i < THDIM * FDIM) bih[i] = (__bf16)wih[i];
    int j = i - THDIM * FDIM;
    if (j >= 0 && j < THDIM * HDIM) bhh[j] = (__bf16)whh[j];
}

// ---------------------------------------------------------------------------
// Message passing: out[dst] += in[src], feature-vectorized (float4 per thread)
// ---------------------------------------------------------------------------
__global__ void scatter_add(const float* __restrict__ in, const int* __restrict__ src,
                            const int* __restrict__ dst, float* __restrict__ out, int E) {
    int idx = blockIdx.x * blockDim.x + threadIdx.x;
    int e = idx >> 5;          // 32 threads per edge
    int f = (idx & 31) << 2;   // 4 floats per thread -> 128 features
    if (e >= E) return;
    int s = src[e];
    int d = dst[e];
    const float4 v = *reinterpret_cast<const float4*>(in + (size_t)s * FDIM + f);
    float* o = out + (size_t)d * FDIM + f;
    atomicAdd(o + 0, v.x);
    atomicAdd(o + 1, v.y);
    atomicAdd(o + 2, v.z);
    atomicAdd(o + 3, v.w);
}

// ---------------------------------------------------------------------------
// B-fragment loader from bf16 weights in global (L2-resident), ISA 16-bit
// layout: lanes 0-15 K = {kb..kb+7, kb+16..kb+23}; lanes 16-31 complementary.
// ---------------------------------------------------------------------------
__device__ __forceinline__ v16bf load_b_frag(const __bf16* __restrict__ base, int ld,
                                             int row, int kb, int half) {
    const __bf16* p = base + (size_t)row * ld + kb + (half ? 8 : 0);
    union { v16bf v; uint4 u[2]; } t;
    t.u[0] = *reinterpret_cast<const uint4*>(p);       // K lo octet (16B)
    t.u[1] = *reinterpret_cast<const uint4*>(p + 16);  // K hi octet (16B)
    return t.v;
}

// A-fragment from LDS (bf16, padded row-major): two ds_load_b128
__device__ __forceinline__ v16bf lds_a_frag(const __bf16* p) {
    union { v16bf v; uint4 u[2]; } t;
    t.u[0] = *reinterpret_cast<const uint4*>(p);
    t.u[1] = *reinterpret_cast<const uint4*>(p + 16);
    return t.v;
}

__device__ __forceinline__ float sigmoidf_(float x) {
    return 1.f / (1.f + __expf(-x));
}

// pack 4 f32 -> 4 bf16 and store as one 8-byte LDS write
__device__ __forceinline__ void st_lds_bf16x4(__bf16* d, float4 v) {
    union { __bf16 h[4]; uint2 u; } t;
    t.h[0] = (__bf16)v.x; t.h[1] = (__bf16)v.y;
    t.h[2] = (__bf16)v.z; t.h[3] = (__bf16)v.w;
    *reinterpret_cast<uint2*>(d) = t.u;
}

#define WMMA_BF16(A, B, C) \
    __builtin_amdgcn_wmma_f32_16x16x32_bf16(false, (A), false, (B), (short)0, (C), false, false)

// ---------------------------------------------------------------------------
// One fused GRU step. Block = 512 threads = 16 waves; block owns a 32-row
// stripe of the [N,256] hidden state; wave w owns column tile w and TWO
// 16-row register tiles (B fragments reused across both -> fewer loads/WMMA).
// ---------------------------------------------------------------------------
__global__ void __launch_bounds__(512)
gru_step(const float* __restrict__ xt, const float* __restrict__ hprev,
         float* __restrict__ hout,
         const __bf16* __restrict__ wbih, const __bf16* __restrict__ wbhh,
         const float* __restrict__ bih, const float* __restrict__ bhh, int N) {
    __shared__ __bf16 sx[MROWS * FPAD];   // 8.5 KB
    __shared__ __bf16 sh[MROWS * HPAD];   // 16.5 KB

    const int tid  = threadIdx.x;
    const int lane = tid & 31;
    const int wave = tid >> 5;            // 0..15 -> column tile
    const int half = lane >> 4;
    const int lm   = lane & 15;

    const int rowBase = blockIdx.x * MROWS;
    const int c0      = wave * 16;

    // ---- cooperative staging: x tile (32x128) -> LDS bf16, 2 float4/thread
#pragma unroll
    for (int rep = 0; rep < 2; ++rep) {
        int i    = (tid + rep * 512) << 2;
        int row  = i >> 7;
        int k    = i & 127;
        int arow = rowBase + row; if (arow > N - 1) arow = N - 1;
        float4 v = *reinterpret_cast<const float4*>(xt + (size_t)arow * FDIM + k);
        st_lds_bf16x4(&sx[row * FPAD + k], v);
    }
    // ---- h tile (32x256) -> LDS bf16, 4 float4/thread
#pragma unroll
    for (int rep = 0; rep < 4; ++rep) {
        int i    = (tid + rep * 512) << 2;
        int row  = i >> 8;
        int k    = i & 255;
        int arow = rowBase + row; if (arow > N - 1) arow = N - 1;
        float4 v = *reinterpret_cast<const float4*>(hprev + (size_t)arow * HDIM + k);
        st_lds_bf16x4(&sh[row * HPAD + k], v);
    }
    __syncthreads();

    v8f accR0  = {}, accR1  = {};  // ir + hr
    v8f accZ0  = {}, accZ1  = {};  // iz + hz
    v8f accIN0 = {}, accIN1 = {};  // inn
    v8f accHN0 = {}, accHN1 = {};  // hn

    const int klofs = half ? 8 : 0;

    // ---- input GEMM: K = 128, A from LDS, B reused across both M tiles ----
#pragma unroll
    for (int kb = 0; kb < FDIM; kb += 32) {
        v16bf a0 = lds_a_frag(&sx[lm * FPAD + kb + klofs]);
        v16bf a1 = lds_a_frag(&sx[(16 + lm) * FPAD + kb + klofs]);
        v16bf bR = load_b_frag(wbih, FDIM, c0 + lm, kb, half);
        v16bf bZ = load_b_frag(wbih, FDIM, HDIM + c0 + lm, kb, half);
        v16bf bN = load_b_frag(wbih, FDIM, 2 * HDIM + c0 + lm, kb, half);
        accR0  = WMMA_BF16(a0, bR, accR0);
        accR1  = WMMA_BF16(a1, bR, accR1);
        accZ0  = WMMA_BF16(a0, bZ, accZ0);
        accZ1  = WMMA_BF16(a1, bZ, accZ1);
        accIN0 = WMMA_BF16(a0, bN, accIN0);
        accIN1 = WMMA_BF16(a1, bN, accIN1);
    }

    // ---- hidden GEMM: K = 256, A from LDS, B reused across both M tiles ----
#pragma unroll
    for (int kb = 0; kb < HDIM; kb += 32) {
        v16bf a0 = lds_a_frag(&sh[lm * HPAD + kb + klofs]);
        v16bf a1 = lds_a_frag(&sh[(16 + lm) * HPAD + kb + klofs]);
        v16bf bR = load_b_frag(wbhh, HDIM, c0 + lm, kb, half);
        v16bf bZ = load_b_frag(wbhh, HDIM, HDIM + c0 + lm, kb, half);
        v16bf bN = load_b_frag(wbhh, HDIM, 2 * HDIM + c0 + lm, kb, half);
        accR0  = WMMA_BF16(a0, bR, accR0);
        accR1  = WMMA_BF16(a1, bR, accR1);
        accZ0  = WMMA_BF16(a0, bZ, accZ0);
        accZ1  = WMMA_BF16(a1, bZ, accZ1);
        accHN0 = WMMA_BF16(a0, bN, accHN0);
        accHN1 = WMMA_BF16(a1, bN, accHN1);
    }

    // ---- fused GRU epilogue (f32 exact), two 16-row tiles ----
    const int col = c0 + lm;
    const float brz = bih[col] + bhh[col];
    const float bzz = bih[HDIM + col] + bhh[HDIM + col];
    const float bin = bih[2 * HDIM + col];
    const float bhn = bhh[2 * HDIM + col];

#pragma unroll
    for (int r = 0; r < 8; ++r) {
        int row = rowBase + r + half * 8;          // tile 0
        if (row < N) {
            float hp = hprev[(size_t)row * HDIM + col];
            float rg = sigmoidf_(accR0[r] + brz);
            float zg = sigmoidf_(accZ0[r] + bzz);
            float ng = tanhf(accIN0[r] + bin + rg * (accHN0[r] + bhn));
            hout[(size_t)row * HDIM + col] = (1.f - zg) * ng + zg * hp;
        }
        int row1 = row + 16;                       // tile 1
        if (row1 < N) {
            float hp = hprev[(size_t)row1 * HDIM + col];
            float rg = sigmoidf_(accR1[r] + brz);
            float zg = sigmoidf_(accZ1[r] + bzz);
            float ng = tanhf(accIN1[r] + bin + rg * (accHN1[r] + bhn));
            hout[(size_t)row1 * HDIM + col] = (1.f - zg) * ng + zg * hp;
        }
    }
}

// ---------------------------------------------------------------------------
// Host launch
// ---------------------------------------------------------------------------
extern "C" void kernel_launch(void* const* d_in, const int* in_sizes, int n_in,
                              void* d_out, int out_size, void* d_ws, size_t ws_size,
                              hipStream_t stream) {
    const float* x    = (const float*)d_in[0];
    const int*   ei   = (const int*)d_in[1];
    const float* Wih  = (const float*)d_in[2];
    const float* Whh  = (const float*)d_in[3];
    const float* bihp = (const float*)d_in[4];
    const float* bhhp = (const float*)d_in[5];

    const int N = in_sizes[0] / FDIM;
    const int E = in_sizes[1] / 2;
    const int* src = ei;
    const int* dst = ei + E;

    // workspace layout
    char* ws = (char*)d_ws;
    const size_t msgBytes = (size_t)N * FDIM * sizeof(float);
    const size_t hBytes   = (size_t)N * HDIM * sizeof(float);
    float* msg[4];
    size_t off = 0;
    for (int i = 0; i < 4; ++i) { msg[i] = (float*)(ws + off); off += msgBytes; }
    float*  hA   = (float*)(ws + off); off += hBytes;
    __bf16* wbih = (__bf16*)(ws + off); off += (size_t)THDIM * FDIM * sizeof(__bf16);
    __bf16* wbhh = (__bf16*)(ws + off); off += (size_t)THDIM * HDIM * sizeof(__bf16);
    (void)ws_size;

    // weights -> bf16 (once per launch; deterministic)
    {
        int n = THDIM * FDIM + THDIM * HDIM;
        cvt_weights<<<(n + 255) / 256, 256, 0, stream>>>(Wih, Whh, wbih, wbhh);
    }

    // message passing: msg_l = scatter_add(msg_{l-1}[src] -> dst), msg_{-1} = x
    {
        const long n4 = (long)N * FDIM / 4;
        const int zgrid = (int)((n4 + 255) / 256);
        const int sgrid = (E * 32 + 255) / 256;
        const float* cur = x;
        for (int l = 0; l < 4; ++l) {
            zero_f32<<<zgrid, 256, 0, stream>>>((float4*)msg[l], n4);
            scatter_add<<<sgrid, 256, 0, stream>>>(cur, src, dst, msg[l], E);
            cur = msg[l];
        }
    }

    // GRU over seq = [msg3(newest), msg2, msg1, msg0, x]; h0 = 0
    {
        const long h4 = (long)N * HDIM / 4;
        zero_f32<<<(int)((h4 + 255) / 256), 256, 0, stream>>>((float4*)hA, h4);

        float* dout = (float*)d_out;
        const float* seq[5] = { msg[3], msg[2], msg[1], msg[0], x };
        const float* hin = hA;
        dim3 grid((N + MROWS - 1) / MROWS);
        for (int s = 0; s < 5; ++s) {
            float* ho = (s % 2 == 0) ? dout : hA;  // parity -> step 4 lands in d_out
            gru_step<<<grid, 512, 0, stream>>>(seq[s], hin, ho, wbih, wbhh, bihp, bhhp, N);
            hin = ho;
        }
    }
}
